// Customize_GCN_82403242541810
// MI455X (gfx1250) — matrix operand order repaired and checked by reading, not compiled, not run on hardware
//
#include <hip/hip_runtime.h>
#include <hip/hip_bf16.h>
#include <math.h>

typedef __attribute__((ext_vector_type(16))) __bf16 v16bf;
typedef __attribute__((ext_vector_type(8)))  __bf16 v8bf;
typedef __attribute__((ext_vector_type(8)))  float  v8f;
typedef __attribute__((__vector_size__(4 * sizeof(int)))) int v4i; // builtin's pointee type

#define D 128          // feature dim (all hidden layers)
#define NCLS 40        // output classes
#define MTILE 128      // rows of X per block in GEMM

// CDNA5 async global->LDS copy path (ASYNCcnt-tracked), with safe fallback.
#if __has_builtin(__builtin_amdgcn_global_load_async_to_lds_b128)
#define USE_ASYNC_LDS 1
#else
#define USE_ASYNC_LDS 0
#endif

// ---------------------------------------------------------------------------
// Fill a float buffer with -inf (segment_max identity; every node has a
// self-loop so no segment stays empty).
// ---------------------------------------------------------------------------
__global__ void fill_neginf_kernel(float* __restrict__ p, long n4) {
    long i = (long)blockIdx.x * blockDim.x + threadIdx.x;
    if (i < n4) {
        float4 v;
        v.x = v.y = v.z = v.w = -INFINITY;
        ((float4*)p)[i] = v;
    }
}

// ---------------------------------------------------------------------------
// H = X @ W   (X: [nrows,128] f32, W: [128,128] f32, H: [nrows,128] f32)
// bf16 WMMA (v_wmma_f32_16x16x32_bf16), f32 accumulate.
// Block = 256 threads (8 waves). Each block: 128 rows x 128 cols.
//  - X tile staged to LDS as raw f32 via global_load_async_to_lds_b128.
//  - W staged transposed as bf16 (converted in VGPRs).
//  - Each wave: 16-row strip; A fragments converted to bf16 once in regs;
//    loops over 8 N-tiles with all 4 B fragments prefetched per tile.
// ---------------------------------------------------------------------------
__global__ void gemm128_wmma_bf16_kernel(const float* __restrict__ X,
                                         const float* __restrict__ W,
                                         float* __restrict__ H,
                                         int nrows) {
    __shared__ float  Xsf[MTILE * D];  // f32 X tile (64 KB), async-staged
    __shared__ __bf16 Wt[D * D];       // W transposed bf16: Wt[n][k] (32 KB)

    const int tid = threadIdx.x;

    // ---- Stage X tile: 4096 x 16B chunks across 256 threads -------------
    #pragma unroll
    for (int it = 0; it < 16; ++it) {
        int i4  = it * 256 + tid;          // 0..4095
        int row = i4 >> 5;                 // 32 float4 per row
        int c4  = (i4 & 31) << 2;
        long grow = (long)blockIdx.x * MTILE + row;
        if (grow >= nrows) grow = nrows - 1;   // clamp: padded rows never stored
#if USE_ASYNC_LDS
        __builtin_amdgcn_global_load_async_to_lds_b128(
            (v4i*)(X + grow * D + c4),         // global source
            (v4i*)&Xsf[row * D + c4],          // LDS destination
            0, 0);
#else
        *(float4*)&Xsf[row * D + c4] = *(const float4*)(X + grow * D + c4);
#endif
    }

    // ---- Stage W transposed (f32 -> bf16): Wt[n*128 + k] = W[k*128 + n] --
    #pragma unroll
    for (int it = 0; it < 16; ++it) {
        int i4 = it * 256 + tid;
        int k  = i4 >> 5;
        int c4 = (i4 & 31) << 2;
        float4 w = *(const float4*)(W + k * D + c4);
        Wt[(c4 + 0) * D + k] = (__bf16)w.x;
        Wt[(c4 + 1) * D + k] = (__bf16)w.y;
        Wt[(c4 + 2) * D + k] = (__bf16)w.z;
        Wt[(c4 + 3) * D + k] = (__bf16)w.w;
    }

#if USE_ASYNC_LDS
#if __has_builtin(__builtin_amdgcn_s_wait_asynccnt)
    __builtin_amdgcn_s_wait_asynccnt(0);
#else
    asm volatile("s_wait_asynccnt 0x0" ::: "memory");
#endif
#endif
    __syncthreads();

    const int wave  = tid >> 5;
    const int lane  = tid & 31;
    const int mbase = wave << 4;       // 16-row strip per wave
    const int r     = lane & 15;
    const int hi    = lane >> 4;       // 0: low K half-group, 1: high

    // ---- Build A fragments once (f32 LDS -> bf16 regs), reused 8x --------
    // A layout (16-bit 16x32): lanes 0-15 hold K {0..7,16..23} of row r,
    // lanes 16-31 hold K {8..15,24..31}.
    v16bf afr[4];
    const float* rowp = &Xsf[(mbase + r) * D];
    #pragma unroll
    for (int kk = 0; kk < 4; ++kk) {
        const int k0 = kk << 5;
        float4 c0a = *(const float4*)(rowp + k0 + hi * 8);
        float4 c0b = *(const float4*)(rowp + k0 + hi * 8 + 4);
        float4 c1a = *(const float4*)(rowp + k0 + 16 + hi * 8);
        float4 c1b = *(const float4*)(rowp + k0 + 16 + hi * 8 + 4);
        afr[kk][0]  = (__bf16)c0a.x; afr[kk][1]  = (__bf16)c0a.y;
        afr[kk][2]  = (__bf16)c0a.z; afr[kk][3]  = (__bf16)c0a.w;
        afr[kk][4]  = (__bf16)c0b.x; afr[kk][5]  = (__bf16)c0b.y;
        afr[kk][6]  = (__bf16)c0b.z; afr[kk][7]  = (__bf16)c0b.w;
        afr[kk][8]  = (__bf16)c1a.x; afr[kk][9]  = (__bf16)c1a.y;
        afr[kk][10] = (__bf16)c1a.z; afr[kk][11] = (__bf16)c1a.w;
        afr[kk][12] = (__bf16)c1b.x; afr[kk][13] = (__bf16)c1b.y;
        afr[kk][14] = (__bf16)c1b.z; afr[kk][15] = (__bf16)c1b.w;
    }

    const long gm0 = (long)blockIdx.x * MTILE + mbase;

    for (int nt = 0; nt < 8; ++nt) {
        const int nbase = nt << 4;

        // Prefetch all 4 B fragments (8x ds_load_b128, single wait).
        // B layout (16-bit 32x16): lanes 0-15 hold col n, K {0..7,16..23};
        // lanes 16-31 hold col n, K {8..15,24..31} (from transposed Wt).
        v16bf bfr[4];
        #pragma unroll
        for (int kk = 0; kk < 4; ++kk) {
            const int k0 = kk << 5;
            const v8bf b0 = *(const v8bf*)&Wt[(nbase + r) * D + k0 + hi * 8];
            const v8bf b1 = *(const v8bf*)&Wt[(nbase + r) * D + k0 + 16 + hi * 8];
            #pragma unroll
            for (int i = 0; i < 8; ++i) { bfr[kk][i] = b0[i]; bfr[kk][i + 8] = b1[i]; }
        }

        v8f acc = {};
        #pragma unroll
        for (int kk = 0; kk < 4; ++kk) {
            acc = __builtin_amdgcn_wmma_f32_16x16x32_bf16(
                false, afr[kk], false, bfr[kk], (short)0, acc, false, false);
        }

        // D layout: lanes 0-15 -> (M=v, N=lane); lanes 16-31 -> (M=v+8, N=lane-16)
        if (gm0 + 16 <= nrows) {                 // wave-uniform fast path
            float* p = H + (gm0 + hi * 8) * D + nbase + r;
            #pragma unroll
            for (int v = 0; v < 8; ++v) p[(long)v * D] = acc[v];
        } else {
            #pragma unroll
            for (int v = 0; v < 8; ++v) {
                long grow = gm0 + v + hi * 8;
                if (grow < nrows) H[grow * D + nbase + r] = acc[v];
            }
        }
    }
}

// ---------------------------------------------------------------------------
// Scatter-max: one wave32 per edge (incl. N self-loop edges appended after E).
// lane owns features [4*lane, 4*lane+4): float4 gather of h[src], scale by
// norm[e], native f32 max atomics (global_atomic_max_num_f32) into agg[dst].
// ---------------------------------------------------------------------------
__global__ void scatter_max_kernel(const float* __restrict__ H,
                                   const int* __restrict__ ei,  // [2,E] flat
                                   const float* __restrict__ norm,
                                   float* __restrict__ agg,
                                   int E, int N) {
    const long gtid = (long)blockIdx.x * blockDim.x + threadIdx.x;
    const long e    = gtid >> 5;         // edge id, uniform per wave
    const int  lane = (int)(gtid & 31);
    if (e >= (long)E + N) return;        // whole-wave exit

    int s, d;
    if (e < E) { s = ei[e]; d = ei[(long)E + e]; }
    else       { s = d = (int)(e - E); }
    const float nv = norm[e];

    const float4 hv = *(const float4*)(H + (size_t)s * D + lane * 4);
    float4 m;
    m.x = hv.x * nv; m.y = hv.y * nv; m.z = hv.z * nv; m.w = hv.w * nv;

    float* p = agg + (size_t)d * D + lane * 4;
    asm volatile("global_atomic_max_num_f32 %0, %1, off"           :: "v"(p), "v"(m.x) : "memory");
    asm volatile("global_atomic_max_num_f32 %0, %1, off offset:4"  :: "v"(p), "v"(m.y) : "memory");
    asm volatile("global_atomic_max_num_f32 %0, %1, off offset:8"  :: "v"(p), "v"(m.z) : "memory");
    asm volatile("global_atomic_max_num_f32 %0, %1, off offset:12" :: "v"(p), "v"(m.w) : "memory");
}

// ---------------------------------------------------------------------------
// x = relu(agg + b), in place on agg.
// ---------------------------------------------------------------------------
__global__ void bias_relu_kernel(float* __restrict__ agg,
                                 const float* __restrict__ b,
                                 long total) {
    long i = (long)blockIdx.x * blockDim.x + threadIdx.x;
    if (i < total) {
        float v = agg[i] + b[(int)(i & (D - 1))];
        agg[i] = v > 0.f ? v : 0.f;
    }
}

// ---------------------------------------------------------------------------
// out = log_softmax(x @ Wout + bout).  One wave32 per node.
// ---------------------------------------------------------------------------
__global__ void out_logsoftmax_kernel(const float* __restrict__ X,
                                      const float* __restrict__ Wout,  // [128,40]
                                      const float* __restrict__ bout,  // [40]
                                      float* __restrict__ out,
                                      int N) {
    __shared__ float lg[8][NCLS];
    const int tid  = threadIdx.x;
    const int wave = tid >> 5;
    const int lane = tid & 31;
    const long node = (long)blockIdx.x * 8 + wave;
    if (node >= N) return;   // whole-wave exit

    const float4 xv = *(const float4*)(X + node * D + lane * 4);
    const int k = lane * 4;
    for (int c = 0; c < NCLS; ++c) {
        float p = xv.x * Wout[(k + 0) * NCLS + c]
                + xv.y * Wout[(k + 1) * NCLS + c]
                + xv.z * Wout[(k + 2) * NCLS + c]
                + xv.w * Wout[(k + 3) * NCLS + c];
        #pragma unroll
        for (int off = 16; off > 0; off >>= 1) p += __shfl_xor(p, off, 32);
        if (lane == 0) lg[wave][c] = p + bout[c];
    }
    // stable log-sum-exp (redundant per lane; 40 LDS reads, cheap)
    float mx = -INFINITY;
    for (int c = 0; c < NCLS; ++c) mx = fmaxf(mx, lg[wave][c]);
    float se = 0.f;
    for (int c = 0; c < NCLS; ++c) se += __expf(lg[wave][c] - mx);
    const float logZ = mx + __logf(se);
    for (int c = lane; c < NCLS; c += 32)
        out[node * NCLS + c] = lg[wave][c] - logZ;
}

// ---------------------------------------------------------------------------
// Driver
// ---------------------------------------------------------------------------
extern "C" void kernel_launch(void* const* d_in, const int* in_sizes, int n_in,
                              void* d_out, int out_size, void* d_ws, size_t ws_size,
                              hipStream_t stream) {
    const float* x    = (const float*)d_in[0];
    const int*   ei   = (const int*)  d_in[1];
    const float* norm = (const float*)d_in[2];
    const float* Ws[3] = { (const float*)d_in[3], (const float*)d_in[5], (const float*)d_in[7] };
    const float* bs[3] = { (const float*)d_in[4], (const float*)d_in[6], (const float*)d_in[8] };
    const float* Wout = (const float*)d_in[9];
    const float* bout = (const float*)d_in[10];
    float* out = (float*)d_out;

    const int N = in_sizes[0] / D;       // 100000
    const int E = in_sizes[1] / 2;       // 1600000
    const long total = (long)N * D;

    float* h   = (float*)d_ws;                  // [N,128]
    float* agg = (float*)d_ws + total;          // [N,128]

    const int gemm_blocks = (N + MTILE - 1) / MTILE;
    const long fill_blocks = (total / 4 + 255) / 256;
    const long scat_blocks = (((long)E + N) * 32 + 255) / 256;
    const long br_blocks   = (total + 255) / 256;
    const int  out_blocks  = (N + 7) / 8;

    const float* xin = x;
    for (int l = 0; l < 3; ++l) {
        gemm128_wmma_bf16_kernel<<<gemm_blocks, 256, 0, stream>>>(xin, Ws[l], h, N);
        fill_neginf_kernel<<<(int)fill_blocks, 256, 0, stream>>>(agg, total / 4);
        scatter_max_kernel<<<(int)scat_blocks, 256, 0, stream>>>(h, ei, norm, agg, E, N);
        bias_relu_kernel<<<(int)br_blocks, 256, 0, stream>>>(agg, bs[l], total);
        xin = agg;   // next layer reads activated aggregate in place
    }
    out_logsoftmax_kernel<<<out_blocks, 256, 0, stream>>>(agg, Wout, bout, out, N);
}
